// PointNetSetAbstractor_90924457656322
// MI455X (gfx1250) — matrix operand order repaired and checked by reading, not compiled, not run on hardware
//
#include <hip/hip_runtime.h>
#include <hip/hip_bf16.h>

typedef __attribute__((ext_vector_type(16))) _Float16 v16h;
typedef __attribute__((ext_vector_type(8)))  _Float16 v8h;
typedef __attribute__((ext_vector_type(8)))  float    v8f;

#define B_    16
#define N_    4096
#define S_    1024
#define K_    32
#define D_    64
#define R2_   (0.2f * 0.2f)

// ---------------- workspace layout (bytes) ----------------
// [0,              65536)   fps_idx   : B*S int
// [65536,        2162688)   ball_idx  : B*S*K int
// [2162688,      2199552)   wfrag     : 18432 f16 (WMMA B-fragment layout)
// [2199552,      2200576)   scale     : 256 f32
// [2200576,      2201600)   bias      : 256 f32
#define WS_FPS   0
#define WS_BALL  65536
#define WS_WF    2162688
#define WS_SCALE 2199552
#define WS_BIAS  2200576

// wfrag element offsets (in f16 elements)
#define WF_L0 0      // 3 kchunks * 4 ntiles * 512
#define WF_L1 6144   // 2 * 4 * 512
#define WF_L2 10240  // 2 * 8 * 512
#define WF_TOTAL 18432

// =====================================================================
// Prep: fold BN into per-channel scale/bias; repack weights (f32 ->
// f16) into the WMMA B-matrix fragment layout so each B operand load
// is two contiguous 16B loads per lane.
//   block fb = kchunk*NT + ntile ; element = lane*16 + j
//   K = kchunk*32 + (j>>3)*16 + (lane>>4)*8 + (j&7) ; col = ntile*16 + (lane&15)
// =====================================================================
__global__ void prep_kernel(const float* __restrict__ W0, const float* __restrict__ W1,
                            const float* __restrict__ W2,
                            const float* __restrict__ b0, const float* __restrict__ g0,
                            const float* __restrict__ be0, const float* __restrict__ m0,
                            const float* __restrict__ v0,
                            const float* __restrict__ b1, const float* __restrict__ g1,
                            const float* __restrict__ be1, const float* __restrict__ m1,
                            const float* __restrict__ v1,
                            const float* __restrict__ b2, const float* __restrict__ g2,
                            const float* __restrict__ be2, const float* __restrict__ m2,
                            const float* __restrict__ v2,
                            _Float16* __restrict__ wf, float* __restrict__ scale,
                            float* __restrict__ bias) {
  int t = blockIdx.x * blockDim.x + threadIdx.x;
  if (t < WF_TOTAL) {
    int u = t; const float* W; int Cin, Cout, NT;
    if (u < WF_L1)      { W = W0; Cin = 67; Cout = 64;  NT = 4; }
    else if (u < WF_L2) { W = W1; Cin = 64; Cout = 64;  NT = 4; u -= WF_L1; }
    else                { W = W2; Cin = 64; Cout = 128; NT = 8; u -= WF_L2; }
    int fb = u >> 9, r = u & 511, lane = r >> 4, j = r & 15;
    int kc = fb / NT, nt = fb % NT;
    int K   = kc * 32 + (j >> 3) * 16 + (lane >> 4) * 8 + (j & 7);
    int col = nt * 16 + (lane & 15);
    wf[t] = (_Float16)((K < Cin) ? W[K * Cout + col] : 0.0f);
  } else if (t < WF_TOTAL + 256) {
    int c = t - WF_TOTAL;
    const float *bp, *gp, *bep, *mp, *vp; int ci;
    if (c < 64)       { bp=b0; gp=g0; bep=be0; mp=m0; vp=v0; ci=c; }
    else if (c < 128) { bp=b1; gp=g1; bep=be1; mp=m1; vp=v1; ci=c-64; }
    else              { bp=b2; gp=g2; bep=be2; mp=m2; vp=v2; ci=c-128; }
    float s = gp[ci] * rsqrtf(vp[ci] + 1e-5f);
    scale[c] = s;
    bias[c]  = (bp[ci] - mp[ci]) * s + bep[ci];
  }
}

// =====================================================================
// Farthest point sampling: one block per batch, min_d lives in LDS,
// sequential 1024-step argmax (first-occurrence tie-break like argmax).
// =====================================================================
__global__ void __launch_bounds__(256) fps_kernel(const float* __restrict__ xyz,
                                                  int* __restrict__ fps_idx) {
  __shared__ float mind[N_];
  __shared__ float rv[256];
  __shared__ int   ri[256];
  __shared__ int   s_last;
  const int b = blockIdx.x;
  const int t = threadIdx.x;
  const float* xb = xyz + (size_t)b * N_ * 3;

  for (int p = t; p < N_; p += 256) mind[p] = 1e10f;
  if (t == 0) { s_last = 0; fps_idx[b * S_] = 0; }
  __syncthreads();

  for (int it = 0; it < S_ - 1; ++it) {
    int last = s_last;
    float cx = xb[last * 3 + 0], cy = xb[last * 3 + 1], cz = xb[last * 3 + 2];
    // contiguous range per thread -> global tie-break by smallest index works
    float bestv = -1.0f; int besti = 0;
    int base = t * (N_ / 256);
    for (int q = 0; q < N_ / 256; ++q) {
      int p = base + q;
      float dx = xb[p * 3 + 0] - cx;
      float dy = xb[p * 3 + 1] - cy;
      float dz = xb[p * 3 + 2] - cz;
      float d = dx * dx + dy * dy + dz * dz;
      float md = mind[p];
      md = (d < md) ? d : md;
      mind[p] = md;
      if (md > bestv) { bestv = md; besti = p; }
    }
    rv[t] = bestv; ri[t] = besti;
    __syncthreads();
    for (int off = 128; off > 0; off >>= 1) {
      if (t < off) {
        float ov = rv[t + off]; int oi = ri[t + off];
        if (ov > rv[t] || (ov == rv[t] && oi < ri[t])) { rv[t] = ov; ri[t] = oi; }
      }
      __syncthreads();
    }
    if (t == 0) { int w = ri[0]; fps_idx[b * S_ + it + 1] = w; s_last = w; }
    __syncthreads();
  }
}

// =====================================================================
// Ball query: one wave per (b,s). wave32 ballot + prefix-popcount gives
// "first K hits in index order" exactly; pad with -1. Also emits new_xyz.
// =====================================================================
__global__ void __launch_bounds__(32) ball_query_kernel(const float* __restrict__ xyz,
                                                        const int* __restrict__ fps_idx,
                                                        int* __restrict__ ball_idx,
                                                        float* __restrict__ new_xyz) {
  const int gid  = blockIdx.x;          // b*S + s
  const int b    = gid >> 10;
  const int lane = threadIdx.x;
  const float* xb = xyz + (size_t)b * N_ * 3;
  const int ci = fps_idx[gid];
  const float cx = xb[ci * 3 + 0], cy = xb[ci * 3 + 1], cz = xb[ci * 3 + 2];
  if (lane == 0) {
    new_xyz[(size_t)gid * 3 + 0] = cx;
    new_xyz[(size_t)gid * 3 + 1] = cy;
    new_xyz[(size_t)gid * 3 + 2] = cz;
  }
  int* out = ball_idx + (size_t)gid * K_;
  int cnt = 0;
  for (int c = 0; c < N_ / 32 && cnt < K_; ++c) {
    int p = c * 32 + lane;
    float dx = xb[p * 3 + 0] - cx;
    float dy = xb[p * 3 + 1] - cy;
    float dz = xb[p * 3 + 2] - cz;
    bool hit = (dx * dx + dy * dy + dz * dz) < R2_;
    unsigned mask = (unsigned)__ballot(hit);
    int rank = __popc(mask & ((1u << lane) - 1u));
    if (hit && (cnt + rank) < K_) out[cnt + rank] = p;
    cnt += __popc(mask);
  }
  if (cnt > K_) cnt = K_;
  if (lane >= cnt) out[lane] = -1;
}

// =====================================================================
// Fused gather + 3x(1x1conv+BN+ReLU) + max-over-K, one wave per (b,s),
// all GEMMs via v_wmma_f32_16x16x32_f16.
// =====================================================================
__device__ __forceinline__ v16h load_a_frag(const _Float16* buf, int stride,
                                            int mtile, int kchunk, int lane) {
  // ISA 16-bit A 16x32 layout: lane m = lane&15; K runs half*8+[0..7] and 16+half*8+[0..7]
  const int m = lane & 15, half = lane >> 4;
  const _Float16* p = buf + (mtile * 16 + m) * stride + kchunk * 32 + half * 8;
  v8h lo = *(const v8h*)p;
  v8h hi = *(const v8h*)(p + 16);
  v16h a;
#pragma unroll
  for (int j = 0; j < 8; ++j) { a[j] = lo[j]; a[j + 8] = hi[j]; }
  return a;
}

__device__ __forceinline__ v16h load_b_frag(const _Float16* wf, int fb, int lane) {
  const v8h* p = (const v8h*)(wf + fb * 512 + lane * 16);
  v8h lo = p[0], hi = p[1];
  v16h w;
#pragma unroll
  for (int j = 0; j < 8; ++j) { w[j] = lo[j]; w[j + 8] = hi[j]; }
  return w;
}

template <int KCHUNKS, int NTILES, int IN_STRIDE, int OUT_STRIDE>
__device__ __forceinline__ void mlp_layer(const _Float16* __restrict__ in,
                                          const _Float16* __restrict__ wf,
                                          const float* __restrict__ sc,
                                          const float* __restrict__ bi,
                                          _Float16* __restrict__ out, int lane) {
  const int half = lane >> 4, nl = lane & 15;
#pragma unroll
  for (int nt = 0; nt < NTILES; ++nt) {
    const int col = nt * 16 + nl;
    const float s = sc[col], bb = bi[col];
#pragma unroll
    for (int mt = 0; mt < 2; ++mt) {
      v8f acc = {};
#pragma unroll
      for (int kc = 0; kc < KCHUNKS; ++kc) {
        v16h a = load_a_frag(in, IN_STRIDE, mt, kc, lane);
        v16h w = load_b_frag(wf, kc * NTILES + nt, lane);
        acc = __builtin_amdgcn_wmma_f32_16x16x32_f16(false, a, false, w,
                                                     (short)0, acc, false, false);
      }
#pragma unroll
      for (int i = 0; i < 8; ++i) {  // C layout: M = mt*16 + 8*half + i, N = col
        float v = acc[i] * s + bb;
        v = v > 0.0f ? v : 0.0f;
        out[(mt * 16 + half * 8 + i) * OUT_STRIDE + col] = (_Float16)v;
      }
    }
  }
}

__global__ void __launch_bounds__(32)
fused_mlp_kernel(const float* __restrict__ xyz, const float* __restrict__ points,
                 const int* __restrict__ ball_idx, const float* __restrict__ new_xyz,
                 const _Float16* __restrict__ wf, const float* __restrict__ scale,
                 const float* __restrict__ bias, float* __restrict__ new_points) {
  __shared__ _Float16 bufA[32 * 96];   // gathered X (67 pad 96) / layer2 out
  __shared__ _Float16 bufB[32 * 64];   // layer1 out
  const int gid  = blockIdx.x;         // b*S + s
  const int b    = gid >> 10;
  const int lane = threadIdx.x;        // also the neighbor row k

  // ---- gather row k = lane: [xyz - center | point feats], pad to 96 ----
  const float cx = new_xyz[(size_t)gid * 3 + 0];
  const float cy = new_xyz[(size_t)gid * 3 + 1];
  const float cz = new_xyz[(size_t)gid * 3 + 2];
  const int id = ball_idx[(size_t)gid * K_ + lane];
  const float* xb = xyz + (size_t)b * N_ * 3;
  const float* pb = points + (size_t)b * N_ * D_;
  float px = 0.0f, py = 0.0f, pz = 0.0f;            // invalid -> grouped_xyz = 0
  if (id >= 0) { px = xb[id * 3 + 0]; py = xb[id * 3 + 1]; pz = xb[id * 3 + 2]; }
  const int src = (id >= 0) ? id : (N_ - 1);        // torch -1 wraps to N-1
  _Float16* row = bufA + lane * 96;
  row[0] = (_Float16)(px - cx);
  row[1] = (_Float16)(py - cy);
  row[2] = (_Float16)(pz - cz);
  const float* f = pb + (size_t)src * D_;
#pragma unroll
  for (int c = 0; c < D_; ++c) row[3 + c] = (_Float16)f[c];
#pragma unroll
  for (int c = 67; c < 96; ++c) row[c] = (_Float16)0.0f;
  __syncthreads();

  // ---- layer 1: (32x96pad) @ (96x64) -> bufB ----
  mlp_layer<3, 4, 96, 64>(bufA, wf + WF_L0, scale + 0, bias + 0, bufB, lane);
  __syncthreads();
  // ---- layer 2: (32x64) @ (64x64) -> bufA ----
  mlp_layer<2, 4, 64, 96>(bufB, wf + WF_L1, scale + 64, bias + 64, bufA, lane);
  __syncthreads();

  // ---- layer 3: (32x64) @ (64x128), fused max over the 32 rows ----
  float* outp = new_points + (size_t)gid * 128;
  const int half = lane >> 4, nl = lane & 15;
#pragma unroll
  for (int nt = 0; nt < 8; ++nt) {
    const int col = nt * 16 + nl;
    const float s = scale[128 + col], bb = bias[128 + col];
    float rmax = 0.0f;                 // post-ReLU values are >= 0
#pragma unroll
    for (int mt = 0; mt < 2; ++mt) {
      v8f acc = {};
#pragma unroll
      for (int kc = 0; kc < 2; ++kc) {
        v16h a = load_a_frag(bufA, 96, mt, kc, lane);
        v16h w = load_b_frag(wf + WF_L2, kc * 8 + nt, lane);
        acc = __builtin_amdgcn_wmma_f32_16x16x32_f16(false, a, false, w,
                                                     (short)0, acc, false, false);
      }
#pragma unroll
      for (int i = 0; i < 8; ++i) {
        float v = acc[i] * s + bb;
        v = v > 0.0f ? v : 0.0f;
        rmax = v > rmax ? v : rmax;
      }
    }
    // lanes l and l+16 hold the same column for different row halves
    float o = __shfl_xor(rmax, 16, 32);
    rmax = o > rmax ? o : rmax;
    if (lane < 16) outp[col] = rmax;
  }
}

// =====================================================================
extern "C" void kernel_launch(void* const* d_in, const int* in_sizes, int n_in,
                              void* d_out, int out_size, void* d_ws, size_t ws_size,
                              hipStream_t stream) {
  (void)in_sizes; (void)n_in; (void)out_size; (void)ws_size;
  const float* xyz    = (const float*)d_in[0];
  const float* points = (const float*)d_in[1];
  const float* W0 = (const float*)d_in[2];
  const float* b0 = (const float*)d_in[3];
  const float* g0 = (const float*)d_in[4];
  const float* be0 = (const float*)d_in[5];
  const float* m0 = (const float*)d_in[6];
  const float* v0 = (const float*)d_in[7];
  const float* W1 = (const float*)d_in[8];
  const float* b1 = (const float*)d_in[9];
  const float* g1 = (const float*)d_in[10];
  const float* be1 = (const float*)d_in[11];
  const float* m1 = (const float*)d_in[12];
  const float* v1 = (const float*)d_in[13];
  const float* W2 = (const float*)d_in[14];
  const float* b2 = (const float*)d_in[15];
  const float* g2 = (const float*)d_in[16];
  const float* be2 = (const float*)d_in[17];
  const float* m2 = (const float*)d_in[18];
  const float* v2 = (const float*)d_in[19];

  char* ws = (char*)d_ws;
  int*      fps_idx  = (int*)(ws + WS_FPS);
  int*      ball_idx = (int*)(ws + WS_BALL);
  _Float16* wfrag    = (_Float16*)(ws + WS_WF);
  float*    scale    = (float*)(ws + WS_SCALE);
  float*    biasv    = (float*)(ws + WS_BIAS);

  float* new_xyz    = (float*)d_out;                 // (B,S,3)
  float* new_points = new_xyz + (size_t)B_ * S_ * 3; // (B,S,128)

  prep_kernel<<<(WF_TOTAL + 256 + 255) / 256, 256, 0, stream>>>(
      W0, W1, W2, b0, g0, be0, m0, v0, b1, g1, be1, m1, v1,
      b2, g2, be2, m2, v2, wfrag, scale, biasv);
  fps_kernel<<<B_, 256, 0, stream>>>(xyz, fps_idx);
  ball_query_kernel<<<B_ * S_, 32, 0, stream>>>(xyz, fps_idx, ball_idx, new_xyz);
  fused_mlp_kernel<<<B_ * S_, 32, 0, stream>>>(xyz, points, ball_idx, new_xyz,
                                               wfrag, scale, biasv, new_points);
}
